// CausalSelfAttention_30700426231933
// MI455X (gfx1250) — matrix operand order repaired and checked by reading, not compiled, not run on hardware
//
#include <hip/hip_runtime.h>
#include <hip/hip_bf16.h>

// Problem constants
constexpr int Bn = 4, Sn = 2048, Dn = 1024, Hn = 16, HDn = 64;

typedef __attribute__((ext_vector_type(16))) __bf16 v16bf;
typedef __attribute__((ext_vector_type(8)))  __bf16 v8bf;
typedef __attribute__((ext_vector_type(8)))  float  v8f;

typedef __attribute__((address_space(3))) __bf16 lds_bf16;
#define LDS_PTR(p) ((lds_bf16*)(p))

union V16U { v16bf v; v8bf h[2]; };

// ---- CDNA5 async copy global->LDS (tracked by ASYNCcnt) ----
__device__ inline void async_b128(const __bf16* g, lds_bf16* l) {
  // VDST = 32-bit LDS byte address, VADDR = 64-bit global address (GV mode)
  asm volatile("global_load_async_to_lds_b128 %0, %1, off" :: "v"(l), "v"(g) : "memory");
}
__device__ inline void wait_async_0()  { asm volatile("s_wait_asynccnt 0x0"  ::: "memory"); }
__device__ inline void wait_async_2()  { asm volatile("s_wait_asynccnt 0x2"  ::: "memory"); }
__device__ inline void wait_async_16() { asm volatile("s_wait_asynccnt 0x10" ::: "memory"); }

// A-operand (16xK tile, rows = M): lane<16 -> K 0..7 then 16..23; lane>=16 -> 8..15 then 24..31
__device__ inline v16bf load_a(const __bf16* base, int ld, int row, int k0, int lane) {
  int lm = lane & 15, hf = (lane >> 4) & 1;
  const __bf16* p = base + (size_t)(row + lm) * ld + k0 + hf * 8;
  V16U u;
  u.h[0] = *(const v8bf*)p;
  u.h[1] = *(const v8bf*)(p + 16);
  return u.v;
}

// B-operand (Kx16 tile, stored transposed: rows = N): lane<16 -> K 0..7 then 16..23; lane>=16 -> 8..15 then 24..31
__device__ inline v16bf load_b(const __bf16* base, int ld, int col, int k0, int lane) {
  int lm = lane & 15, hf = (lane >> 4) & 1;
  const __bf16* p = base + (size_t)(col + lm) * ld + k0 + hf * 8;
  V16U u;
  u.h[0] = *(const v8bf*)p;
  u.h[1] = *(const v8bf*)(p + 16);
  return u.v;
}

__device__ inline v8f wmma_bf16(v16bf a, v16bf b, v8f c) {
  return __builtin_amdgcn_wmma_f32_16x16x32_bf16(false, a, false, b, (short)0, c, false, false);
}

// ---------------- conversion kernels ----------------
__global__ void cvt_bf16_kernel(const float* __restrict__ in, __bf16* __restrict__ out, int n) {
  int i = blockIdx.x * blockDim.x + threadIdx.x;
  if (i < n) out[i] = (__bf16)in[i];
}

// in: R x C (f32, row-major) -> out: C x R (bf16, row-major)  i.e. transposed copy
__global__ void cvt_t_bf16_kernel(const float* __restrict__ in, __bf16* __restrict__ out, int R, int C) {
  int i = blockIdx.x * blockDim.x + threadIdx.x;
  if (i < R * C) {
    int r = i / C, c = i % C;
    out[(size_t)c * R + r] = (__bf16)in[i];
  }
}

// ======================= shared GEMM core =======================
// C(128x64 per block) = A(row-major bf16, ld=Dn) @ B^T(stored N x K row-major).
// Weight tile (64 cols x 64 K = 8KB) async-staged to LDS, explicit double buffer:
// loop body handles buffer 0 then buffer 1 with compile-time buffer indices so the
// accumulators flow through a single symmetric wmma chain (no register rotation).
template <typename EPILOGUE>
__device__ inline void gemm_core(const __bf16* __restrict__ Am, const __bf16* __restrict__ wT,
                                 __bf16 (*ldsB)[64 * 64], EPILOGUE epi) {
  const int t = threadIdx.x;
  const int lane = t & 31;
  const int wave = t >> 5;
  const int lm = lane & 15, hf = (lane >> 4) & 1;
  const int rowbase = blockIdx.x * 128 + wave * 16;
  const int colbase = blockIdx.y * 64;

  // staging: 64 rows x 128B = 8KB; 256 threads x 2 chunks of 16B
  const int srow = t >> 2;            // 0..63
  const int sel  = (t & 3) * 8;       // element offset of first chunk

  auto stage = [&](int buf, int k0) {
#pragma unroll
    for (int c = 0; c < 2; ++c) {
      const __bf16* g = wT + (size_t)(colbase + srow) * Dn + k0 + sel + c * 32;
      async_b128(g, LDS_PTR(&ldsB[buf][srow * 64 + sel + c * 32]));
    }
  };

  v8f acc[4] = {};

  auto compute = [&](const __bf16* Bt, int k0) {
    v16bf a0 = load_a(Am, Dn, rowbase, k0, lane);
    v16bf a1 = load_a(Am, Dn, rowbase, k0 + 32, lane);
    v16bf b0 = load_b(Bt, 64,  0, 0, lane);
    v16bf b1 = load_b(Bt, 64, 16, 0, lane);
    v16bf b2 = load_b(Bt, 64, 32, 0, lane);
    v16bf b3 = load_b(Bt, 64, 48, 0, lane);
    acc[0] = wmma_bf16(a0, b0, acc[0]);
    acc[1] = wmma_bf16(a0, b1, acc[1]);
    acc[2] = wmma_bf16(a0, b2, acc[2]);
    acc[3] = wmma_bf16(a0, b3, acc[3]);
    b0 = load_b(Bt, 64,  0, 32, lane);
    b1 = load_b(Bt, 64, 16, 32, lane);
    b2 = load_b(Bt, 64, 32, 32, lane);
    b3 = load_b(Bt, 64, 48, 32, lane);
    acc[0] = wmma_bf16(a1, b0, acc[0]);
    acc[1] = wmma_bf16(a1, b1, acc[1]);
    acc[2] = wmma_bf16(a1, b2, acc[2]);
    acc[3] = wmma_bf16(a1, b3, acc[3]);
  };

  stage(0, 0);
  for (int k0 = 0; k0 < Dn; k0 += 128) {
    stage(1, k0 + 64);                    // k0+64 <= Dn-64 always (Dn % 128 == 0)
    wait_async_2();                       // buffer 0 landed (async retires in order)
    __syncthreads();
    compute((const __bf16*)ldsB[0], k0);
    __syncthreads();                      // everyone done reading buf0

    if (k0 + 128 < Dn) { stage(0, k0 + 128); wait_async_2(); }
    else               { wait_async_0(); }
    __syncthreads();
    compute((const __bf16*)ldsB[1], k0 + 64);
    __syncthreads();
  }

#pragma unroll
  for (int j = 0; j < 4; ++j)
#pragma unroll
    for (int r = 0; r < 8; ++r)
      epi(rowbase + r + hf * 8, colbase + j * 16 + lm, acc[j][r]);
}

// ---------------- QKV projection: (B*S x D) @ (D x 3D) ----------------
__global__ __launch_bounds__(256) void qkv_gemm_kernel(
    const __bf16* __restrict__ xb, const __bf16* __restrict__ wT,
    const float* __restrict__ bqkv,
    __bf16* __restrict__ qb, __bf16* __restrict__ kb, __bf16* __restrict__ vTb) {
  __shared__ __align__(16) __bf16 ldsB[2][64 * 64];
  gemm_core(xb, wT, ldsB, [&](int row, int col, float v) {
    float val = v + bqkv[col];
    int b = row / Sn, s = row % Sn;
    int which = col / Dn, d = col % Dn;
    int h = d >> 6, hd = d & 63;
    size_t hbs = (size_t)(b * Hn + h) * Sn + s;
    if (which == 0)      qb[hbs * HDn + hd] = (__bf16)(val * 0.125f);  // fold 1/sqrt(64)
    else if (which == 1) kb[hbs * HDn + hd] = (__bf16)val;
    else                 vTb[((size_t)(b * Hn + h) * HDn + hd) * Sn + s] = (__bf16)val;
  });
}

// ---------------- output projection: (B*S x D) @ (D x D) + bo -> f32 ----------------
__global__ __launch_bounds__(256) void out_gemm_kernel(
    const __bf16* __restrict__ ab, const __bf16* __restrict__ wT,
    const float* __restrict__ bo, float* __restrict__ out) {
  __shared__ __align__(16) __bf16 ldsB[2][64 * 64];
  gemm_core(ab, wT, ldsB, [&](int row, int col, float v) {
    out[(size_t)row * Dn + col] = v + bo[col];
  });
}

// ---------------- flash attention: one wave per (b, h, 16-query tile) ----------------
// K/V tiles for each 32-key block async-copied into LDS; explicit double-buffer pair loop
// so the copy of block i+1 overlaps WMMAs + softmax of block i without acc register churn.
__global__ __launch_bounds__(32) void attn_kernel(
    const __bf16* __restrict__ qb, const __bf16* __restrict__ kb,
    const __bf16* __restrict__ vTb, __bf16* __restrict__ aout) {
  __shared__ __align__(16) __bf16 ldsK[2][32 * 64];  // [key][hd]
  __shared__ __align__(16) __bf16 ldsV[2][64 * 32];  // [hd][key]
  __shared__ __align__(32) __bf16 lds_p[16 * 32];    // P tile relayout buffer

  const int lane = threadIdx.x & 31;
  const int lm = lane & 15, hf = (lane >> 4) & 1;
  const int id = blockIdx.x;
  const int qt = id & 127;         // S/16 = 128 query tiles
  const int h  = (id >> 7) & 15;
  const int b  = id >> 11;
  const int qbase = qt * 16;

  const __bf16* qh = qb + (size_t)(b * Hn + h) * Sn * HDn;
  const __bf16* kh = kb + (size_t)(b * Hn + h) * Sn * HDn;
  const __bf16* vh = vTb + (size_t)(b * Hn + h) * HDn * Sn;

  // ALiBi: slope_h = (2^8)^(-(h+1)/H) = 2^(-0.5*(h+1)) for H=16
  const float slope = exp2f(-0.5f * (float)(h + 1));

  const v16bf qa0 = load_a(qh, HDn, qbase, 0, lane);
  const v16bf qa1 = load_a(qh, HDn, qbase, 32, lane);

  // stage one 32-key block: K 32x64 (8 async b128) + V 64x32 (8 async b128)
  auto stage = [&](int buf, int kpos) {
#pragma unroll
    for (int c = 0; c < 8; ++c) {
      const __bf16* g = kh + (size_t)(kpos + lane) * HDn + c * 8;
      async_b128(g, LDS_PTR(&ldsK[buf][lane * 64 + c * 8]));
    }
#pragma unroll
    for (int c = 0; c < 8; ++c) {
      int vrow = lane + 32 * (c >> 2);
      const __bf16* g = vh + (size_t)vrow * Sn + kpos + (c & 3) * 8;
      async_b128(g, LDS_PTR(&ldsV[buf][vrow * 32 + (c & 3) * 8]));
    }
  };

  v8f acc[4] = {};
  float mrow[8], lrow[8];
#pragma unroll
  for (int r = 0; r < 8; ++r) { mrow[r] = -1e30f; lrow[r] = 0.f; }

  auto compute = [&](const __bf16* Kt, const __bf16* Vt, int kpos) {
    // scores: S = Q @ K^T  (two 16x16 tiles over 32 keys); hoist all B-operand loads
    v16bf kb0 = load_b(Kt, 64,  0,  0, lane);
    v16bf kb1 = load_b(Kt, 64,  0, 32, lane);
    v16bf kb2 = load_b(Kt, 64, 16,  0, lane);
    v16bf kb3 = load_b(Kt, 64, 16, 32, lane);
    v8f s0 = {}, s1 = {};
    s0 = wmma_bf16(qa0, kb0, s0);
    s1 = wmma_bf16(qa0, kb2, s1);
    s0 = wmma_bf16(qa1, kb1, s0);
    s1 = wmma_bf16(qa1, kb3, s1);

    // online softmax with ALiBi + causal mask; row stats per C-layout half-wave
#pragma unroll
    for (int r = 0; r < 8; ++r) {
      int qrow = qbase + r + hf * 8;
      int c0 = kpos + lm, c1 = kpos + 16 + lm;
      float v0 = (c0 <= qrow) ? (s0[r] + slope * (float)(c0 - qrow)) : -1e30f;
      float v1 = (c1 <= qrow) ? (s1[r] + slope * (float)(c1 - qrow)) : -1e30f;
      float sm = fmaxf(v0, v1);
#pragma unroll
      for (int o = 1; o < 16; o <<= 1) sm = fmaxf(sm, __shfl_xor(sm, o, 32));
      float mnew = fmaxf(mrow[r], sm);
      float p0 = __expf(v0 - mnew);
      float p1 = __expf(v1 - mnew);
      float rs = p0 + p1;
#pragma unroll
      for (int o = 1; o < 16; o <<= 1) rs += __shfl_xor(rs, o, 32);
      float corr = __expf(mrow[r] - mnew);
      lrow[r] = lrow[r] * corr + rs;
      mrow[r] = mnew;
      acc[0][r] *= corr; acc[1][r] *= corr; acc[2][r] *= corr; acc[3][r] *= corr;
      // stage P (bf16) in LDS to convert C-layout -> A-operand layout
      lds_p[(r + hf * 8) * 32 + lm]      = (__bf16)p0;
      lds_p[(r + hf * 8) * 32 + 16 + lm] = (__bf16)p1;
    }
    __syncthreads();

    // O += P @ V  (K-dim = 32 keys; V tile in LDS is [hd][key], contiguous B-operand)
    v16bf pa  = load_a(lds_p, 32, 0, 0, lane);
    v16bf vb0 = load_b(Vt, 32,  0, 0, lane);
    v16bf vb1 = load_b(Vt, 32, 16, 0, lane);
    v16bf vb2 = load_b(Vt, 32, 32, 0, lane);
    v16bf vb3 = load_b(Vt, 32, 48, 0, lane);
    acc[0] = wmma_bf16(pa, vb0, acc[0]);
    acc[1] = wmma_bf16(pa, vb1, acc[1]);
    acc[2] = wmma_bf16(pa, vb2, acc[2]);
    acc[3] = wmma_bf16(pa, vb3, acc[3]);
    __syncthreads();
  };

  const int nb = (qbase + 16 + 31) >> 5;   // key blocks of 32 needed (causal)
  stage(0, 0);
  for (int kblk = 0; kblk < nb; kblk += 2) {
    const int kpos0 = kblk * 32;
    const bool has1 = (kblk + 1 < nb);
    if (has1) { stage(1, kpos0 + 32); wait_async_16(); }
    else      { wait_async_0(); }
    compute((const __bf16*)ldsK[0], (const __bf16*)ldsV[0], kpos0);
    if (has1) {
      if (kblk + 2 < nb) { stage(0, kpos0 + 64); wait_async_16(); }
      else               { wait_async_0(); }
      compute((const __bf16*)ldsK[1], (const __bf16*)ldsV[1], kpos0 + 32);
    }
  }

  // normalize and emit attention output [b, s, h*64+hd] as bf16 for the out-proj GEMM
#pragma unroll
  for (int j = 0; j < 4; ++j) {
#pragma unroll
    for (int r = 0; r < 8; ++r) {
      int qrow = qbase + r + hf * 8;
      int d = h * HDn + j * 16 + lm;
      float val = acc[j][r] / lrow[r];
      aout[((size_t)b * Sn + qrow) * Dn + d] = (__bf16)val;
    }
  }
}

extern "C" void kernel_launch(void* const* d_in, const int* in_sizes, int n_in,
                              void* d_out, int out_size, void* d_ws, size_t ws_size,
                              hipStream_t stream) {
  const float* x    = (const float*)d_in[0];
  // d_in[1] = mask (unused; causal mask computed analytically)
  const float* Wqkv = (const float*)d_in[2];
  const float* bqkv = (const float*)d_in[3];
  const float* Wo   = (const float*)d_in[4];
  const float* bo   = (const float*)d_in[5];
  float* out = (float*)d_out;

  char* ws = (char*)d_ws;
  size_t off = 0;
  auto alloc = [&](size_t bytes) -> void* {
    void* p = ws + off;
    off += (bytes + 255) & ~(size_t)255;
    return p;
  };
  const size_t nTok = (size_t)Bn * Sn;            // 8192
  __bf16* xb   = (__bf16*)alloc(nTok * Dn * 2);            // x in bf16
  __bf16* wqT  = (__bf16*)alloc((size_t)3 * Dn * Dn * 2);  // Wqkv^T bf16 (3072 x 1024)
  __bf16* woT  = (__bf16*)alloc((size_t)Dn * Dn * 2);      // Wo^T bf16
  __bf16* qb   = (__bf16*)alloc(nTok * Dn * 2);            // Q [b,h,s,hd] (pre-scaled)
  __bf16* kb   = (__bf16*)alloc(nTok * Dn * 2);            // K [b,h,s,hd]
  __bf16* vTb  = (__bf16*)alloc(nTok * Dn * 2);            // V [b,h,hd,s]
  __bf16* aout = (__bf16*)alloc(nTok * Dn * 2);            // attention output [b,s,d]

  const int nX = (int)(nTok * Dn);
  cvt_bf16_kernel<<<(nX + 255) / 256, 256, 0, stream>>>(x, xb, nX);
  cvt_t_bf16_kernel<<<(Dn * 3 * Dn + 255) / 256, 256, 0, stream>>>(Wqkv, wqT, Dn, 3 * Dn);
  cvt_t_bf16_kernel<<<(Dn * Dn + 255) / 256, 256, 0, stream>>>(Wo, woT, Dn, Dn);

  qkv_gemm_kernel<<<dim3((unsigned)(nTok / 128), 3 * Dn / 64), 256, 0, stream>>>(
      xb, wqT, bqkv, qb, kb, vTb);

  attn_kernel<<<Bn * Hn * (Sn / 16), 32, 0, stream>>>(qb, kb, vTb, aout);

  out_gemm_kernel<<<dim3((unsigned)(nTok / 128), Dn / 64), 256, 0, stream>>>(
      aout, woT, bo, out);
}
